// LowRankAttention_1821066133994
// MI455X (gfx1250) — compile-verified, matrix operand order checked by reading, not run on hardware
//
#include <hip/hip_runtime.h>
#include <hip/hip_bf16.h>

// ---------------- problem constants ----------------
constexpr int BATCH = 4, SEQ = 2048, DMODEL = 1024, NH = 16, HD = 64, RK = 32;
constexpr int ROWS = BATCH * SEQ; // 8192

// ---------------- workspace layout (bytes) ----------------
constexpr size_t O_T   = 0;                                  // bf16 [ROWS][RK]
constexpr size_t SZ_T  = (size_t)ROWS * RK * 2;
constexpr size_t O_WQ  = O_T + SZ_T;                         // bf16 [NH][RK(n)][RK(k)]  (transposed)
constexpr size_t SZ_W  = (size_t)NH * RK * RK * 2;
constexpr size_t O_WK  = O_WQ + SZ_W;
constexpr size_t O_WV  = O_WK + SZ_W;
constexpr size_t O_BQ  = O_WV + SZ_W;                        // f32 [NH][RK]
constexpr size_t SZ_B  = (size_t)NH * RK * 4;
constexpr size_t O_BK  = O_BQ + SZ_B;
constexpr size_t O_BV  = O_BK + SZ_B;
constexpr size_t O_M   = O_BV + SZ_B;                        // bf16 [NH][RK(n)][RK(k)]  (transposed)
constexpr size_t O_OV  = O_M + SZ_W;                         // bf16 [DMODEL(n)][RK(k)]  (transposed)
constexpr size_t SZ_OV = (size_t)RK * DMODEL * 2;
constexpr size_t O_QL  = O_OV + SZ_OV;                       // bf16 [B][H][S][RK]
constexpr size_t SZ_L  = (size_t)BATCH * NH * SEQ * RK * 2;
constexpr size_t O_KL  = O_QL + SZ_L;                        // bf16 [B][H][S][RK]
constexpr size_t O_VL  = O_KL + SZ_L;                        // bf16 [B][H][RK][S]  (transposed!)
constexpr size_t O_CTX = O_VL + SZ_L;                        // f32 [B][H][S][RK]

// ---------------- WMMA types / helpers ----------------
typedef __attribute__((ext_vector_type(16))) __bf16 v16bf;
typedef __attribute__((ext_vector_type(8)))  __bf16 v8bf;
typedef __attribute__((ext_vector_type(8)))  float  v8f;

// Native f32 -> bf16 convert (hardware v_cvt, RTNE)
__device__ __forceinline__ __bf16 f2bf(float f) { return (__bf16)f; }

__device__ __forceinline__ v8f wmma_bf16(v16bf a, v16bf b, v8f c) {
  // D(16x16 f32) = A(16x32 bf16) x B(32x16 bf16) + C
  return __builtin_amdgcn_wmma_f32_16x16x32_bf16(false, a, false, b,
                                                 (short)0, c, false, false);
}

// Operand from a contiguous 32-element bf16 run (A row / B column when the
// source is stored K-contiguous). khalf = lane/16 selects the lane's K-halves
// per the ISA 16-bit operand layout. Two 16B-aligned 128-bit loads.
__device__ __forceinline__ v16bf load_frag_row(const __bf16* row, int khalf) {
  v8bf lo = *(const v8bf*)(row + khalf * 8);
  v8bf hi = *(const v8bf*)(row + 16 + khalf * 8);
  return __builtin_shufflevector(lo, hi, 0, 1, 2, 3, 4, 5, 6, 7,
                                 8, 9, 10, 11, 12, 13, 14, 15);
}

// ---------------- K0: precompose low-rank weights (stored N-major) ----------------
__global__ __launch_bounds__(256) void k0_precompose(
    const float* __restrict__ qkv_v, const float* __restrict__ qkv_b,
    const float* __restrict__ u_attn, const float* __restrict__ v_attn,
    const float* __restrict__ out_u, const float* __restrict__ out_v,
    char* __restrict__ ws) {
  __bf16* WqT = (__bf16*)(ws + O_WQ);
  __bf16* WkT = (__bf16*)(ws + O_WK);
  __bf16* WvT = (__bf16*)(ws + O_WV);
  float*  bq  = (float*)(ws + O_BQ);
  float*  bk  = (float*)(ws + O_BK);
  float*  bv  = (float*)(ws + O_BV);
  __bf16* MhT = (__bf16*)(ws + O_M);
  __bf16* OVT = (__bf16*)(ws + O_OV);
  const float scale = rsqrtf((float)RK);
  int tid = threadIdx.x;

  for (int idx = tid; idx < NH * RK * RK; idx += 256) {
    int h = idx / (RK * RK), r1 = (idx / RK) % RK, r2 = idx % RK;
    float sq = 0.f, sk = 0.f, sv = 0.f, sm = 0.f;
    for (int d = 0; d < HD; ++d) {
      float u = u_attn[(size_t)(h * HD + d) * RK + r2];
      sq += qkv_v[(size_t)r1 * (3 * DMODEL) + 0 * DMODEL + h * HD + d] * u;
      sk += qkv_v[(size_t)r1 * (3 * DMODEL) + 1 * DMODEL + h * HD + d] * u;
      sv += qkv_v[(size_t)r1 * (3 * DMODEL) + 2 * DMODEL + h * HD + d] * u;
      sm += v_attn[(size_t)(h * RK + r1) * HD + d] *
            out_u[(size_t)(h * HD + d) * RK + r2];
    }
    // transposed store: [h][r2 (=N)][r1 (=K)]
    size_t tix = ((size_t)h * RK + r2) * RK + r1;
    WqT[tix] = f2bf(sq * scale);
    WkT[tix] = f2bf(sk);
    WvT[tix] = f2bf(sv);
    MhT[tix] = f2bf(sm);
  }
  for (int idx = tid; idx < NH * RK; idx += 256) {
    int h = idx / RK, r2 = idx % RK;
    float sq = 0.f, sk = 0.f, sv = 0.f;
    for (int d = 0; d < HD; ++d) {
      float u = u_attn[(size_t)(h * HD + d) * RK + r2];
      sq += qkv_b[0 * DMODEL + h * HD + d] * u;
      sk += qkv_b[1 * DMODEL + h * HD + d] * u;
      sv += qkv_b[2 * DMODEL + h * HD + d] * u;
    }
    bq[idx] = sq * scale; bk[idx] = sk; bv[idx] = sv;
  }
  // out_v transposed: OVT[n][r]
  for (int idx = tid; idx < RK * DMODEL; idx += 256) {
    int r = idx / DMODEL, n = idx % DMODEL;
    OVT[(size_t)n * RK + r] = f2bf(out_v[idx]);
  }
}

// ---------------- K1: t = x @ qkv_u  (HBM-bound, reads x once) ----------------
__global__ __launch_bounds__(256) void k1_proj_t(const float* __restrict__ x,
                                                 const float* __restrict__ qkv_u,
                                                 char* __restrict__ ws) {
  __shared__ float xt[8][DMODEL]; // 32 KB
  int tid = threadIdx.x;
  int rb = blockIdx.x * 8;
  for (int i = tid; i < 8 * DMODEL; i += 256)
    xt[i / DMODEL][i % DMODEL] = x[(size_t)rb * DMODEL + i];
  __syncthreads();
  int rl = tid / RK, r = tid % RK;
  float acc = 0.f;
  for (int d = 0; d < DMODEL; ++d) acc += xt[rl][d] * qkv_u[(size_t)d * RK + r];
  ((__bf16*)(ws + O_T))[(size_t)(rb + rl) * RK + r] = f2bf(acc);
}

// ---------------- K2: q_low/k_low/v_low = t @ W{q,k,v}[h] (WMMA) ----------------
__global__ __launch_bounds__(256) void k2_lowproj(char* __restrict__ ws) {
  const __bf16* T   = (const __bf16*)(ws + O_T);
  const __bf16* WqT = (const __bf16*)(ws + O_WQ);
  const __bf16* WkT = (const __bf16*)(ws + O_WK);
  const __bf16* WvT = (const __bf16*)(ws + O_WV);
  const float*  bq  = (const float*)(ws + O_BQ);
  const float*  bk  = (const float*)(ws + O_BK);
  const float*  bv  = (const float*)(ws + O_BV);
  __bf16* QL  = (__bf16*)(ws + O_QL);
  __bf16* KL  = (__bf16*)(ws + O_KL);
  __bf16* VLT = (__bf16*)(ws + O_VL);

  int lane = threadIdx.x & 31, wave = threadIdx.x >> 5;
  int khalf = lane >> 4, rl = lane & 15;
  int rowbase = blockIdx.x * 128 + wave * 16;
  int b = rowbase / SEQ, s0 = rowbase % SEQ;

  v16bf a = load_frag_row(T + (size_t)(rowbase + rl) * RK, khalf);
  v8f z = {};
  for (int h = 0; h < NH; ++h) {
    const __bf16* wq = WqT + (size_t)h * RK * RK;
    const __bf16* wk = WkT + (size_t)h * RK * RK;
    const __bf16* wv = WvT + (size_t)h * RK * RK;
#pragma unroll
    for (int half = 0; half < 2; ++half) {
      int n = half * 16 + rl;
      v8f cq = wmma_bf16(a, load_frag_row(wq + (size_t)n * RK, khalf), z);
      v8f ck = wmma_bf16(a, load_frag_row(wk + (size_t)n * RK, khalf), z);
      v8f cv = wmma_bf16(a, load_frag_row(wv + (size_t)n * RK, khalf), z);
      float bqv = bq[h * RK + n], bkv = bk[h * RK + n], bvv = bv[h * RK + n];
      // QL/KL: row-major [s][rank] (read-contiguous later), scattered b16 stores
#pragma unroll
      for (int i = 0; i < 8; ++i) {
        size_t base = (((size_t)(b * NH + h)) * SEQ + s0 + i + 8 * khalf) * RK + n;
        QL[base] = f2bf(cq[i] + bqv);
        KL[base] = f2bf(ck[i] + bkv);
      }
      // VL: transposed [rank][s] -> this lane's 8 results are s-contiguous: one b128 store
      v8bf pv;
#pragma unroll
      for (int i = 0; i < 8; ++i) pv[i] = f2bf(cv[i] + bvv);
      *(v8bf*)(VLT + ((size_t)(b * NH + h) * RK + n) * SEQ + s0 + 8 * khalf) = pv;
    }
  }
}

// ---------------- K3: fused flash attention (rank-32 inner dim) ----------------
// One block = (b, h, 128-query tile); each wave owns 16 queries.
// Transposed scores: S^T = K_tile x q_low^T  -> C layout of S^T IS the
// A-operand layout of P for the P x V WMMA (no cross-lane shuffles for P).
// v_low is stored [rank][seq], so PV B-fragments are contiguous b128 loads too.
__global__ __launch_bounds__(256) void k3_attn(const int* __restrict__ mask,
                                               char* __restrict__ ws) {
  __shared__ float maskadd[SEQ]; // 8 KB
  const __bf16* QL  = (const __bf16*)(ws + O_QL);
  const __bf16* KL  = (const __bf16*)(ws + O_KL);
  const __bf16* VLT = (const __bf16*)(ws + O_VL);
  float* CTX = (float*)(ws + O_CTX);

  int blk = blockIdx.x;
  int qt = blk & 15, h = (blk >> 4) & 15, b = blk >> 8;

  for (int i = threadIdx.x; i < SEQ; i += 256)
    maskadd[i] = mask[b * SEQ + i] ? 0.f : -1e30f;
  __syncthreads();

  int lane = threadIdx.x & 31, wave = threadIdx.x >> 5;
  int khalf = lane >> 4, rl = lane & 15;
  int q0 = qt * 128 + wave * 16;

  const __bf16* qlb  = QL + ((size_t)(b * NH + h) * SEQ) * RK;
  const __bf16* klb  = KL + ((size_t)(b * NH + h) * SEQ) * RK;
  const __bf16* vr0  = VLT + ((size_t)(b * NH + h) * RK + rl) * SEQ;       // rank rl
  const __bf16* vr1  = VLT + ((size_t)(b * NH + h) * RK + 16 + rl) * SEQ;  // rank rl+16

  // B operand (invariant): q_low^T, column = this lane's query
  v16bf bqf = load_frag_row(qlb + (size_t)(q0 + rl) * RK, khalf);

  v8f acc0 = {}, acc1 = {};
  float mstat = -__builtin_inff(), lstat = 0.f;
  v8f z = {};

  for (int kb = 0; kb < SEQ; kb += 32) {
    if (kb + 32 < SEQ) {
      __builtin_prefetch(klb + (size_t)(kb + 32 + rl) * RK, 0, 0);
      __builtin_prefetch(vr0 + kb + 32, 0, 0);
    }
    // A operands: 16-key tiles of k_low (contiguous rows, b128 loads)
    v16bf ak0 = load_frag_row(klb + (size_t)(kb + rl) * RK, khalf);
    v16bf ak1 = load_frag_row(klb + (size_t)(kb + 16 + rl) * RK, khalf);
    v8f s0 = wmma_bf16(ak0, bqf, z); // S^T keys kb..kb+15
    v8f s1 = wmma_bf16(ak1, bqf, z); // S^T keys kb+16..kb+31

    // key mask (row = key = i + 8*khalf in this C layout)
#pragma unroll
    for (int i = 0; i < 8; ++i) {
      s0[i] += maskadd[kb + i + 8 * khalf];
      s1[i] += maskadd[kb + 16 + i + 8 * khalf];
    }

    // online softmax, stats per column (query) = per lane (dup across halves)
    float cmax = -__builtin_inff();
#pragma unroll
    for (int i = 0; i < 8; ++i) cmax = fmaxf(cmax, fmaxf(s0[i], s1[i]));
    cmax = fmaxf(cmax, __shfl_xor(cmax, 16, 32));
    float mnew = fmaxf(mstat, cmax);
    float f = __expf(mstat - mnew);

    float psum = 0.f;
    v16bf ap; // P in A-operand layout, directly from S^T C layout
#pragma unroll
    for (int i = 0; i < 8; ++i) {
      float p0 = __expf(s0[i] - mnew);
      float p1 = __expf(s1[i] - mnew);
      psum += p0 + p1;
      ap[i] = f2bf(p0);
      ap[8 + i] = f2bf(p1);
    }
    psum += __shfl_xor(psum, 16, 32);
    lstat = lstat * f + psum;
    mstat = mnew;

    // rescale accumulator: acc row q = i + 8*khalf; f lives at lane q
#pragma unroll
    for (int i = 0; i < 8; ++i) {
      float fr = __shfl(f, i + 8 * khalf, 32);
      acc0[i] *= fr;
      acc1[i] *= fr;
    }

    // ctx(16q x 32r) += P(16q x 32keys) x v_low_chunk(32keys x 32r)
    // B column = rank; VLT rows are key-contiguous -> b128 loads
    v16bf bv0 = load_frag_row(vr0 + kb, khalf);
    v16bf bv1 = load_frag_row(vr1 + kb, khalf);
    acc0 = wmma_bf16(ap, bv0, acc0);
    acc1 = wmma_bf16(ap, bv1, acc1);
  }

  // normalize and store context_low
  float* ctx = CTX + ((size_t)(b * NH + h) * SEQ + q0) * RK;
#pragma unroll
  for (int i = 0; i < 8; ++i) {
    float lr = __shfl(lstat, i + 8 * khalf, 32);
    float inv = (lr > 0.f) ? (1.f / lr) : 0.f;
    ctx[(size_t)(i + 8 * khalf) * RK + rl] = acc0[i] * inv;
    ctx[(size_t)(i + 8 * khalf) * RK + 16 + rl] = acc1[i] * inv;
  }
}

// ---------------- K4: out = (sum_h ctx[h] @ M[h]) @ out_v + out_b ----------------
__global__ __launch_bounds__(256) void k4_out(const float* __restrict__ out_b,
                                              char* __restrict__ ws,
                                              float* __restrict__ out) {
  __shared__ float stage[8][16][RK]; // 16 KB: per-wave C->A relayout buffer
  const float*  CTX = (const float*)(ws + O_CTX);
  const __bf16* MhT = (const __bf16*)(ws + O_M);
  const __bf16* OVT = (const __bf16*)(ws + O_OV);

  int lane = threadIdx.x & 31, wave = threadIdx.x >> 5;
  int khalf = lane >> 4, rl = lane & 15;
  int rowbase = blockIdx.x * 128 + wave * 16;
  int b = rowbase / SEQ, s0 = rowbase % SEQ;

  v8f acc0 = {}, acc1 = {};
  for (int h = 0; h < NH; ++h) {
    const float* ctxr = CTX + ((size_t)(b * NH + h) * SEQ + s0) * RK;
    v16bf a;
#pragma unroll
    for (int e = 0; e < 8; ++e)
      a[e] = f2bf(ctxr[(size_t)rl * RK + khalf * 8 + e]);
#pragma unroll
    for (int e = 0; e < 8; ++e)
      a[8 + e] = f2bf(ctxr[(size_t)rl * RK + 16 + khalf * 8 + e]);
    const __bf16* mh = MhT + (size_t)h * RK * RK;
    acc0 = wmma_bf16(a, load_frag_row(mh + (size_t)rl * RK, khalf), acc0);
    acc1 = wmma_bf16(a, load_frag_row(mh + (size_t)(16 + rl) * RK, khalf), acc1);
  }

  // relayout out_low (C layout -> A layout) through LDS
#pragma unroll
  for (int i = 0; i < 8; ++i) {
    stage[wave][i + 8 * khalf][rl] = acc0[i];
    stage[wave][i + 8 * khalf][16 + rl] = acc1[i];
  }
  __syncthreads();
  v16bf al;
#pragma unroll
  for (int e = 0; e < 8; ++e) al[e] = f2bf(stage[wave][rl][khalf * 8 + e]);
#pragma unroll
  for (int e = 0; e < 8; ++e) al[8 + e] = f2bf(stage[wave][rl][16 + khalf * 8 + e]);

  v8f z = {};
  for (int nt = 0; nt < DMODEL / 16; ++nt) {
    int n = nt * 16 + rl;
    v8f c = wmma_bf16(al, load_frag_row(OVT + (size_t)n * RK, khalf), z);
    float ob = out_b[n];
#pragma unroll
    for (int i = 0; i < 8; ++i)
      out[((size_t)rowbase + i + 8 * khalf) * DMODEL + n] = c[i] + ob;
  }
}

// ---------------- host launcher ----------------
extern "C" void kernel_launch(void* const* d_in, const int* in_sizes, int n_in,
                              void* d_out, int out_size, void* d_ws, size_t ws_size,
                              hipStream_t stream) {
  (void)in_sizes; (void)n_in; (void)out_size; (void)ws_size;
  const float* x     = (const float*)d_in[0];
  const int*   mask  = (const int*)d_in[1];
  const float* qkv_u = (const float*)d_in[2];
  const float* qkv_v = (const float*)d_in[3];
  const float* qkv_b = (const float*)d_in[4];
  const float* u_at  = (const float*)d_in[5];
  const float* v_at  = (const float*)d_in[6];
  const float* out_u = (const float*)d_in[7];
  const float* out_v = (const float*)d_in[8];
  const float* out_b = (const float*)d_in[9];
  char*  ws  = (char*)d_ws;
  float* out = (float*)d_out;

  k0_precompose<<<1, 256, 0, stream>>>(qkv_v, qkv_b, u_at, v_at, out_u, out_v, ws);
  k1_proj_t<<<ROWS / 8, 256, 0, stream>>>(x, qkv_u, ws);
  k2_lowproj<<<ROWS / 128, 256, 0, stream>>>(ws);
  k3_attn<<<BATCH * NH * (SEQ / 128), 256, 0, stream>>>(mask, ws);
  k4_out<<<ROWS / 128, 256, 0, stream>>>(out_b, ws, out);
}
